// LSTM_18416819765873
// MI455X (gfx1250) — matrix operand order repaired and checked by reading, not compiled, not run on hardware
//
#include <hip/hip_runtime.h>
#include <hip/hip_bf16.h>

// ---------------- problem dims ----------------
#define T_STEPS 1024
#define BATCH   64
#define IC      512
#define HC      512
#define NGATE   2048       // 4*HC
#define KTOT    1024       // IC + HC
#define NWG     32         // one WG per 16-wide hc slice
#define NTHREADS 256       // 8 waves (wave32)

// ---------------- LDS layout (bytes) ----------------
#define BOFF 0
#define AOFF (4*32*1024)
#define GOFF (AOFF + 4*32*1024)
#define COFF (GOFF + 4*BATCH*16*4)
#define SMEM_BYTES (COFF + BATCH*16*4)   // 282624 bytes < 320KB LDS/WGP

// ---------------- workspace layout (bytes) ----------------
#define WS_BAR    0                   // barrier counters
#define WS_HSTATE 4096                // packed bf16 h state [2 layers][2 parity][64KB]
#define WS_WPACK  (512ull*1024)       // bf16 weights, fragment order: 8MB
#define WS_XPACK  (16ull<<20)         // bf16 x, fragment order [t][64KB]: 64MB
#define WS_HBUF   (80ull<<20)         // bf16 layer-0 h history, fragment order: 64MB

// ---------------- optional CDNA5 async global->LDS path ----------------
#if defined(__has_builtin)
#  if __has_builtin(__builtin_amdgcn_global_load_async_to_lds_b128)
#    define HAVE_ASYNC_LDS 1
#  endif
#endif
#ifndef HAVE_ASYNC_LDS
#  define HAVE_ASYNC_LDS 0
#endif

typedef __bf16 bf16x16 __attribute__((ext_vector_type(16)));
typedef float  f32x8   __attribute__((ext_vector_type(8)));

#if HAVE_ASYNC_LDS
typedef int int4v __attribute__((ext_vector_type(4)));
typedef __attribute__((address_space(1))) int4v* gptr_t;   // global int4*
typedef __attribute__((address_space(3))) int4v* lptr_t;   // LDS int4*
__device__ __forceinline__ gptr_t to_gbl(const void* p) {
  return (gptr_t)(unsigned long long)p;
}
__device__ __forceinline__ lptr_t to_lds(const void* p) {
  // generic LDS pointer: low 32 bits are the LDS byte offset (aperture scheme)
  return (lptr_t)(unsigned)(unsigned long long)p;
}
__device__ __forceinline__ void wait_async0() {
#if defined(__has_builtin) && __has_builtin(__builtin_amdgcn_s_wait_asynccnt)
  __builtin_amdgcn_s_wait_asynccnt(0);
#else
  asm volatile("s_wait_asynccnt 0x0" ::: "memory");
#endif
}
#endif

__device__ __forceinline__ unsigned f2bf_bits(float f) {
  unsigned u = __float_as_uint(f);
  return (u + 0x7FFFu + ((u >> 16) & 1u)) >> 16;   // round-to-nearest-even
}
__device__ __forceinline__ unsigned pack_bf2(float a, float b) {
  return (f2bf_bits(a) & 0xFFFFu) | (f2bf_bits(b) << 16);
}
__device__ __forceinline__ float sigm(float v) {
  return 1.0f / (1.0f + __expf(-v));
}
__device__ __forceinline__ float tanh_fast(float v) {
  v = fminf(15.0f, fmaxf(-15.0f, v));
  float e = __expf(2.0f * v);
  return (e - 1.0f) / (e + 1.0f);
}
// fragment byte offset for packed h / x panels: element (m, col) of a 64 x 512
// bf16 panel in WMMA-A fragment order [mtile(4)][kblock(16)][lane(32)][32B]
__device__ __forceinline__ size_t frag_off(int m, int col) {
  int kb   = col >> 5;
  int lane = (m & 15) + (((col >> 3) & 1) << 4);
  int kr   = (col & 7) | ((col & 16) >> 1);
  return (size_t)(((((m >> 4) * 16 + kb) << 10) + lane * 32 + (kr << 1)));
}

// ---------------------------------------------------------------------------
// Init: hstate parity0 <- bf16(h0) (fragment-packed); parity1 <- 0; bar <- 0.
// ---------------------------------------------------------------------------
__global__ void lstm_init(const float* __restrict__ h0,
                          char* __restrict__ hstate,
                          unsigned* __restrict__ bar) {
  int e = blockIdx.x * blockDim.x + threadIdx.x;   // 16384 threads
  if (e < 2 * BATCH * HC / 4) {
    int l  = e >> 13;
    int r  = e & 8191;
    int m  = r >> 7;
    int j4 = (r & 127) << 2;
    const float* src = h0 + ((size_t)l * BATCH + m) * HC + j4;
    uint2 hp;
    hp.x = pack_bf2(src[0], src[1]);
    hp.y = pack_bf2(src[2], src[3]);
    size_t off = frag_off(m, j4);
    *(uint2*)(hstate + (size_t)(l * 2 + 0) * 65536 + off) = hp;
    uint2 z; z.x = 0u; z.y = 0u;
    *(uint2*)(hstate + (size_t)(l * 2 + 1) * 65536 + off) = z;
  }
  if (e < 8) bar[e] = 0;
}

// ---------------------------------------------------------------------------
// Pack W = [Wi | Wh] into bf16 WMMA-B fragment order:
// [l][ntile 0..127][kblock 0..31][lane 0..31][16 bf16].
// B 32x16 layout: lane = (n&15) + 16*((k>>3)&1), byte = ((k&7)|((k&16)>>1))*2.
// ---------------------------------------------------------------------------
__global__ void lstm_pack_w(const float* __restrict__ Wi,
                            const float* __restrict__ Wh,
                            char* __restrict__ wpack) {
  long p = (long)blockIdx.x * blockDim.x + threadIdx.x;  // pair index
  if (p >= 2L * NGATE * (KTOT / 2)) return;              // 2,097,152
  int l   = (int)(p >> 20);
  int rem = (int)(p & 1048575);
  int n   = rem >> 9;
  int k   = (rem & 511) << 1;
  float v0, v1;
  if (k < IC) {
    const float* W = Wi + ((size_t)l * NGATE + n) * IC + k;
    v0 = W[0]; v1 = W[1];
  } else {
    const float* W = Wh + ((size_t)l * NGATE + n) * HC + (k - IC);
    v0 = W[0]; v1 = W[1];
  }
  unsigned u  = pack_bf2(v0, v1);
  int ntg     = n >> 4;
  int kb      = k >> 5;
  int lane    = (n & 15) + (((k >> 3) & 1) << 4);
  size_t dst  = ((((size_t)l * 128 + ntg) * 32 + kb) * 32 + lane) * 32 + (((k & 7) | ((k & 16) >> 1)) << 1);
  *(unsigned*)(wpack + dst) = u;
}

// ---------------------------------------------------------------------------
// Pack x (f32 [T][64][512]) into bf16 WMMA-A fragment order, per t: 64KB panel.
// ---------------------------------------------------------------------------
__global__ void lstm_pack_x(const float* __restrict__ x,
                            char* __restrict__ xpack) {
  long p = (long)blockIdx.x * blockDim.x + threadIdx.x;
  if (p >= (long)T_STEPS * BATCH * (IC / 8)) return;     // 4,194,304
  int t   = (int)(p >> 12);
  int rem = (int)(p & 4095);
  int m   = rem >> 6;
  int k8  = (rem & 63) << 3;
  const float4* xp = (const float4*)(x + (((size_t)t * BATCH + m) * IC + k8));
  float4 f0 = xp[0], f1 = xp[1];
  uint4 pk;
  pk.x = pack_bf2(f0.x, f0.y); pk.y = pack_bf2(f0.z, f0.w);
  pk.z = pack_bf2(f1.x, f1.y); pk.w = pack_bf2(f1.z, f1.w);
  int kb   = k8 >> 5;
  int lane = (m & 15) + (((k8 >> 3) & 1) << 4);
  size_t dst = (size_t)t * 65536 + ((((m >> 4) * 16 + kb) << 10) + lane * 32 + (k8 & 16));
  *(uint4*)(xpack + dst) = pk;
}

// ---------------------------------------------------------------------------
// Persistent fused LSTM scan. Grid = 32 WGs x 256 threads (8 waves).
// ---------------------------------------------------------------------------
__global__ __launch_bounds__(NTHREADS, 1)
void lstm_scan(const float* __restrict__ bi,
               const float* __restrict__ bh,
               const float* __restrict__ c0,
               const char*  __restrict__ wpack,
               const char*  __restrict__ xpack,
               char* __restrict__ hstate,
               char* __restrict__ hbufpk,
               float* __restrict__ out,
               unsigned* __restrict__ bar) {
  extern __shared__ char smem[];
  const int tid  = threadIdx.x;
  const int wg   = blockIdx.x;       // hc slice
  const int lane = tid & 31;
  const int wave = tid >> 5;
  const int mt   = wave >> 1;        // m-tile 0..3 (batch rows)
  const int g0   = (wave & 1) * 2;   // this wave: gates g0, g0+1
  const int hi   = lane >> 4;
  const int j16  = lane & 15;

  unsigned* bar_cnt = bar;
  unsigned* bar_gen = bar + 1;
  float* G = (float*)(smem + GOFF);
  float* C = (float*)(smem + COFF);

  // per-thread elementwise coordinates (4 consecutive hc columns each)
  const int em  = tid >> 2;
  const int ej0 = (tid & 3) << 2;
  const int ecol = wg * 16 + ej0;
  const size_t ehoff = frag_off(em, ecol);

  for (int l = 0; l < 2; ++l) {
    __syncthreads();
    // ---- stage this WG's weight slice into LDS (straight 128KB copy) ----
    for (int g = 0; g < 4; ++g) {
      const char* src = wpack + ((size_t)(l * 128 + g * 32 + wg)) * 32768;
      char* dst = smem + BOFF + g * 32768;
      #pragma unroll
      for (int c = 0; c < 8; ++c) {
        int off = (tid + c * NTHREADS) * 16;
        *(uint4*)(dst + off) = *(const uint4*)(src + off);
      }
    }
    // ---- bias (bi+bh) folded into accumulator init ----
    int ncol0   = g0 * HC + wg * 16 + j16;
    float bias0 = bi[l * NGATE + ncol0]      + bh[l * NGATE + ncol0];
    float bias1 = bi[l * NGATE + ncol0 + HC] + bh[l * NGATE + ncol0 + HC];
    // ---- init cell-state slice from c0 ----
    {
      const float4 cc = *(const float4*)(c0 + ((size_t)l * BATCH + em) * HC + ecol);
      *(float4*)&C[tid << 2] = cc;
    }
    __syncthreads();

    const char* xbase = (l == 0) ? xpack : hbufpk;
    for (int t = 0; t < T_STEPS; ++t) {
      const char* xsrc = xbase + (size_t)t * 65536;
      const char* hrd  = hstate + (size_t)(l * 2 + (t & 1)) * 65536;
      char*       hwr  = hstate + (size_t)(l * 2 + ((t + 1) & 1)) * 65536;

      if (t + 1 < T_STEPS)
        __builtin_prefetch(xsrc + 65536 + (tid << 8), 0, 1);

      // ---- stage A panel: identity copies (already fragment-packed) ----
#if HAVE_ASYNC_LDS
      #pragma unroll
      for (int i = 0; i < 16; ++i) {
        int off = (tid + (i << 8)) << 4;
        __builtin_amdgcn_global_load_async_to_lds_b128(
            to_gbl(xsrc + off), to_lds(smem + AOFF + off), 0, 0);
      }
      #pragma unroll
      for (int i = 0; i < 16; ++i) {
        int off = (tid + (i << 8)) << 4;
        __builtin_amdgcn_global_load_async_to_lds_b128(
            to_gbl(hrd + off), to_lds(smem + AOFF + 65536 + off), 0, 0);
      }
      wait_async0();
#else
      #pragma unroll
      for (int i = 0; i < 16; ++i) {
        int off = (tid + (i << 8)) << 4;
        *(uint4*)(smem + AOFF + off) = *(const uint4*)(xsrc + off);
      }
      #pragma unroll
      for (int i = 0; i < 16; ++i) {
        int off = (tid + (i << 8)) << 4;
        *(uint4*)(smem + AOFF + 65536 + off) = *(const uint4*)(hrd + off);
      }
#endif
      __syncthreads();

      // ---- GEMM: two 16x16 gate tiles per wave, K=1024 (64 WMMAs/wave) ----
      f32x8 acc0 = {bias0, bias0, bias0, bias0, bias0, bias0, bias0, bias0};
      f32x8 acc1 = {bias1, bias1, bias1, bias1, bias1, bias1, bias1, bias1};
      const char* AxL = smem + AOFF + (mt << 14) + lane * 32;          // x part
      const char* AhL = AxL + 65536;                                   // h part
      const char* Bb0 = smem + BOFF + (g0 << 15) + lane * 32;
      const char* Bb1 = Bb0 + 32768;
      #pragma unroll
      for (int kb = 0; kb < 16; ++kb) {
        bf16x16 a  = *(const bf16x16*)(AxL + (kb << 10));
        bf16x16 b0 = *(const bf16x16*)(Bb0 + (kb << 10));
        bf16x16 b1 = *(const bf16x16*)(Bb1 + (kb << 10));
        acc0 = __builtin_amdgcn_wmma_f32_16x16x32_bf16(false, a, false, b0,
                                                       (short)0, acc0, false, false);
        acc1 = __builtin_amdgcn_wmma_f32_16x16x32_bf16(false, a, false, b1,
                                                       (short)0, acc1, false, false);
      }
      #pragma unroll
      for (int kb = 0; kb < 16; ++kb) {
        bf16x16 a  = *(const bf16x16*)(AhL + (kb << 10));
        bf16x16 b0 = *(const bf16x16*)(Bb0 + ((16 + kb) << 10));
        bf16x16 b1 = *(const bf16x16*)(Bb1 + ((16 + kb) << 10));
        acc0 = __builtin_amdgcn_wmma_f32_16x16x32_bf16(false, a, false, b0,
                                                       (short)0, acc0, false, false);
        acc1 = __builtin_amdgcn_wmma_f32_16x16x32_bf16(false, a, false, b1,
                                                       (short)0, acc1, false, false);
      }

      // ---- publish gate tiles to LDS (C/D layout: r->M, lane hi adds 8) ----
      #pragma unroll
      for (int r = 0; r < 8; ++r) {
        int m = (mt << 4) + r + (hi << 3);
        G[(g0 * BATCH + m) * 16 + j16]       = acc0[r];
        G[((g0 + 1) * BATCH + m) * 16 + j16] = acc1[r];
      }
      __syncthreads();

      // ---- elementwise LSTM cell update: 4 consecutive hc cols / thread ----
      {
        float4 gf = *(const float4*)&G[(0 * BATCH + em) * 16 + ej0];
        float4 gi = *(const float4*)&G[(1 * BATCH + em) * 16 + ej0];
        float4 go = *(const float4*)&G[(2 * BATCH + em) * 16 + ej0];
        float4 gg = *(const float4*)&G[(3 * BATCH + em) * 16 + ej0];
        float4 cc = *(const float4*)&C[tid << 2];
        float4 cn, hn;
        cn.x = cc.x * sigm(gf.x) + tanh_fast(gg.x) * sigm(gi.x);
        cn.y = cc.y * sigm(gf.y) + tanh_fast(gg.y) * sigm(gi.y);
        cn.z = cc.z * sigm(gf.z) + tanh_fast(gg.z) * sigm(gi.z);
        cn.w = cc.w * sigm(gf.w) + tanh_fast(gg.w) * sigm(gi.w);
        hn.x = sigm(go.x) * tanh_fast(cn.x);
        hn.y = sigm(go.y) * tanh_fast(cn.y);
        hn.z = sigm(go.z) * tanh_fast(cn.z);
        hn.w = sigm(go.w) * tanh_fast(cn.w);
        *(float4*)&C[tid << 2] = cn;

        uint2 hp;
        hp.x = pack_bf2(hn.x, hn.y);
        hp.y = pack_bf2(hn.z, hn.w);
        *(uint2*)(hwr + ehoff) = hp;                     // recurrent state (packed)
        if (l == 0) {
          *(uint2*)(hbufpk + (size_t)t * 65536 + ehoff) = hp;   // layer-1 input panel
        } else {
          *(float4*)(out + ((size_t)t * BATCH + em) * HC + ecol) = hn;
        }
        if (t == T_STEPS - 1) {
          size_t hsb = (size_t)T_STEPS * BATCH * HC;
          *(float4*)(out + hsb + ((size_t)l * BATCH + em) * HC + ecol) = hn;
          *(float4*)(out + hsb + (size_t)2 * BATCH * HC +
                     ((size_t)l * BATCH + em) * HC + ecol) = cn;
        }
      }

      // ---- device-wide barrier: make h_t visible to all 32 WGs ----
      __syncthreads();
      if (tid == 0) {
        __threadfence();
        unsigned gen  = __hip_atomic_load(bar_gen, __ATOMIC_ACQUIRE, __HIP_MEMORY_SCOPE_AGENT);
        unsigned prev = __hip_atomic_fetch_add(bar_cnt, 1u, __ATOMIC_ACQ_REL, __HIP_MEMORY_SCOPE_AGENT);
        if (prev == (unsigned)(gridDim.x - 1)) {
          __hip_atomic_store(bar_cnt, 0u, __ATOMIC_RELAXED, __HIP_MEMORY_SCOPE_AGENT);
          __hip_atomic_fetch_add(bar_gen, 1u, __ATOMIC_ACQ_REL, __HIP_MEMORY_SCOPE_AGENT);
        } else {
          while (__hip_atomic_load(bar_gen, __ATOMIC_ACQUIRE, __HIP_MEMORY_SCOPE_AGENT) == gen) {
            __builtin_amdgcn_s_sleep(2);
          }
        }
        __threadfence();
      }
      __syncthreads();
    }
  }
}

// ---------------------------------------------------------------------------
extern "C" void kernel_launch(void* const* d_in, const int* in_sizes, int n_in,
                              void* d_out, int out_size, void* d_ws, size_t ws_size,
                              hipStream_t stream) {
  (void)in_sizes; (void)n_in; (void)out_size; (void)ws_size;
  const float* x  = (const float*)d_in[0];
  const float* Wi = (const float*)d_in[1];
  const float* bi = (const float*)d_in[2];
  const float* Wh = (const float*)d_in[3];
  const float* bh = (const float*)d_in[4];
  const float* h0 = (const float*)d_in[5];
  const float* c0 = (const float*)d_in[6];
  float* out = (float*)d_out;

  char* ws = (char*)d_ws;
  unsigned* bar   = (unsigned*)(ws + WS_BAR);
  char*    hstate = ws + WS_HSTATE;
  char*    wpack  = ws + WS_WPACK;
  char*    xpack  = ws + WS_XPACK;
  char*    hbufpk = ws + WS_HBUF;

  lstm_init  <<<64,    NTHREADS, 0, stream>>>(h0, hstate, bar);
  lstm_pack_w<<<8192,  NTHREADS, 0, stream>>>(Wi, Wh, wpack);
  lstm_pack_x<<<16384, NTHREADS, 0, stream>>>(x, xpack);
  lstm_scan  <<<NWG, NTHREADS, SMEM_BYTES, stream>>>(bi, bh, c0, wpack, xpack,
                                                     hstate, hbufpk, out, bar);
}
